// Conv4d_8220567404794
// MI455X (gfx1250) — matrix operand order repaired and checked by reading, not compiled, not run on hardware
//
#include <hip/hip_runtime.h>

// ---------------------------------------------------------------------------
// 4D conv [B=2,CIN=64,S=18^4] * W[81,64,64] -> [B=2,COUT=64,16^4]
// Implicit GEMM on bf16 WMMA (v_wmma_f32_16x16x32_bf16), wave32, pipelined.
//  - block = (b,w,x,y-group of 8): M = 128 (8 y x 16 z), N = 64
//  - 512 threads = 16 waves; wave = (mw, nw): two 16x16 M-tiles (y0+mw, y0+mw+4)
//    per N-tile -> each B fragment feeds 2 WMMAs
//  - outer loop over 9 (kw,kx); [10 y][18 z][64 c] slab staged once, ky and kz
//    slid inside LDS -> 36 WMMAs per wave per staging phase
//  - double-buffered LDS (51.8KB); next slab's global loads issued before the
//    WMMA block, converted + stored after -> staging overlapped with compute
//  - staging stores packed 2xbf16 dwords (conflict-free ds_store_b32)
//  - W pre-transposed/converted once to bf16 Wt[off][co][c] in d_ws (663KB, L2)
// ---------------------------------------------------------------------------

namespace {
constexpr int B_    = 2;
constexpr int CIN_  = 64;
constexpr int COUT_ = 64;
constexpr int KD_   = 3;
constexpr int S_    = 18;
constexpr int OS_   = 16;
constexpr int S2_   = S_ * S_;        // 324
constexpr int S3_   = S2_ * S_;       // 5832
constexpr int S4_   = S3_ * S_;       // 104976
constexpr int NOFF_ = KD_ * KD_ * KD_ * KD_;      // 81
constexpr int WELT_ = NOFF_ * CIN_ * COUT_;       // 331776
constexpr int NT_   = KD_ * KD_;      // 9 staging phases (kw,kx)
constexpr int YS_   = 10;             // staged y rows per slab (8 outputs + 2 halo)
constexpr int APAD_ = 72;             // LDS z-row stride in ushorts (144B)
constexpr int NBLK_ = B_ * OS_ * OS_ * 2;         // 1024 blocks (2 y-groups)
}

typedef __attribute__((ext_vector_type(16))) __bf16       v16bf;
typedef __attribute__((ext_vector_type(8)))  float        v8f;
typedef __attribute__((ext_vector_type(4)))  unsigned int u32x4;
typedef __attribute__((ext_vector_type(4)))  float        f32x4;

static __device__ __forceinline__ unsigned int f2bf(float f) {
    // round-to-nearest-even f32 -> bf16 (returned in low 16 bits)
    unsigned u = __float_as_uint(f);
    u += 0x7FFFu + ((u >> 16) & 1u);
    return u >> 16;
}

// Pass 1: Wt[off][co][c] = bf16(kernel[off][c][co])   (663KB, stays in L2)
__global__ void conv4d_wprep(const float* __restrict__ kern,
                             unsigned short* __restrict__ wt) {
    int idx = blockIdx.x * blockDim.x + threadIdx.x;
    if (idx >= WELT_) return;
    int off = idx >> 12;       // / (64*64)
    int r   = idx & 4095;
    int co  = r >> 6;
    int c   = r & 63;
    wt[idx] = (unsigned short)f2bf(kern[off * 4096 + c * 64 + co]);
}

// Pass 2: pipelined implicit-GEMM conv via bf16 WMMA
__global__ __launch_bounds__(512)
void conv4d_wmma(const float* __restrict__ in,
                 const unsigned short* __restrict__ wt,
                 float* __restrict__ out) {
    // double-buffered slab: [buf][y=0..9][z=0..17][c] bf16, z-rows padded to 144B
    __shared__ unsigned short At[2][YS_][S_][APAD_];

    const int tid  = threadIdx.x;
    const int lane = tid & 31;
    const int wave = tid >> 5;           // 0..15
    const int n    = lane & 15;
    const int hi   = lane >> 4;          // 0: lanes 0-15, 1: lanes 16-31
    const int koff = hi << 3;            // A K sub-offset: 0 or 8
    const int boff = hi << 4;            // unused; B loads also use koff (0 or 8)
    const int nw   = wave & 3;
    const int mw   = wave >> 2;          // 0..3: M-tile pair (y0+mw, y0+mw+4)
    const int n0   = nw << 4;            // this wave's co tile base
    const int m    = n;                  // A z-row for this lane

    const int bid = blockIdx.x;
    const int y0 = (bid & 1) << 3;       // y-group base: 0 or 8
    const int x  = (bid >> 1) & 15;
    const int w  = (bid >> 5) & 15;
    const int b  = bid >> 9;

    // staging: waves 0..9 (tid<320): thread -> (dy, c-pair), full 18-z rows
    const int dy = tid >> 5;             // 0..9 when staging-active
    const int cp = tid & 31;             // c pair index: c = 2cp, 2cp+1
    const bool stager = (tid < 32 * YS_);

    const float* inb0 = in + (b * CIN_ + 2 * cp) * S4_ + w * S3_ + x * S2_
                           + (y0 + dy) * S_;

    v8f acc0 = {}, acc1 = {};
    float ra[S_], rb[S_];

    // prologue: stage slab 0 (kw=kx=0) into buffer 0
    if (stager) {
        #pragma unroll
        for (int j = 0; j < S_; ++j) { ra[j] = inb0[j]; rb[j] = inb0[j + S4_]; }
        #pragma unroll
        for (int j = 0; j < S_; ++j)
            *reinterpret_cast<unsigned int*>(&At[0][dy][j][2 * cp]) =
                f2bf(ra[j]) | (f2bf(rb[j]) << 16);
    }

    for (int t = 0; t < NT_; ++t) {
        __syncthreads();   // slab t visible; prior reads of buf[(t+1)&1] done

        // ---- issue global loads for slab t+1 (overlaps with compute below)
        if (t + 1 < NT_ && stager) {
            int t1  = t + 1;
            int kw1 = t1 / 3;
            int kx1 = t1 - kw1 * 3;
            const float* p = inb0 + kw1 * S3_ + kx1 * S2_;
            #pragma unroll
            for (int j = 0; j < S_; ++j) { ra[j] = p[j]; rb[j] = p[j + S4_]; }
        }

        // ---- compute phase t: 3 ky x 3 kz x 2 halves x 2 M-tiles = 36 WMMAs
        {
            const unsigned short (*A)[S_][APAD_] = At[t & 1];
            const int offb = t * 9;      // off = ((kw*3+kx)*3+ky)*3+kz = t*9+ky*3+kz
            #pragma unroll
            for (int ky = 0; ky < 3; ++ky) {
                #pragma unroll
                for (int kz = 0; kz < 3; ++kz) {
                    const unsigned short* wb =
                        wt + (offb + ky * 3 + kz) * 4096 + (n0 + n) * 64;
                    #pragma unroll
                    for (int half = 0; half < 2; ++half) {
                        const int c0 = half << 5;
                        union { u32x4 q[2]; v16bf h; } au0, au1, bu;
                        // B 32x16 bf16: lane col co=n0+n; elems = K c0+koff..+7, c0+16+koff..+7
                        bu.q[0] = *reinterpret_cast<const u32x4*>(wb + c0 + koff);
                        bu.q[1] = *reinterpret_cast<const u32x4*>(wb + c0 + 16 + koff);
                        // A 16x32 bf16, tile e: y-slab mw+4e+ky, z-row m+kz
                        au0.q[0] = *reinterpret_cast<const u32x4*>(
                            &A[mw + ky][m + kz][c0 + koff]);
                        au0.q[1] = *reinterpret_cast<const u32x4*>(
                            &A[mw + ky][m + kz][c0 + 16 + koff]);
                        au1.q[0] = *reinterpret_cast<const u32x4*>(
                            &A[mw + 4 + ky][m + kz][c0 + koff]);
                        au1.q[1] = *reinterpret_cast<const u32x4*>(
                            &A[mw + 4 + ky][m + kz][c0 + 16 + koff]);
                        acc0 = __builtin_amdgcn_wmma_f32_16x16x32_bf16(
                            false, au0.h, false, bu.h, (short)0, acc0, false, false);
                        acc1 = __builtin_amdgcn_wmma_f32_16x16x32_bf16(
                            false, au1.h, false, bu.h, (short)0, acc1, false, false);
                    }
                }
            }
        }

        // ---- convert + store slab t+1 into the other buffer
        if (t + 1 < NT_ && stager) {
            #pragma unroll
            for (int j = 0; j < S_; ++j)
                *reinterpret_cast<unsigned int*>(&At[(t + 1) & 1][dy][j][2 * cp]) =
                    f2bf(ra[j]) | (f2bf(rb[j]) << 16);
        }
    }

    // ---- store: lane l, reg rr -> D[z = rr + 8*hi][co = n0 + n], two y tiles
    const int co  = n0 + n;
    const int zhi = hi << 3;
    #pragma unroll
    for (int e = 0; e < 2; ++e) {
        const int y = y0 + mw + 4 * e;
        float* po = out + (b * COUT_ + co) * (OS_ * OS_ * OS_ * OS_)
                        + w * (OS_ * OS_ * OS_) + x * (OS_ * OS_) + y * OS_ + zhi;
        const v8f& a = e ? acc1 : acc0;
        f32x4 s0 = { a[0], a[1], a[2], a[3] };
        f32x4 s1 = { a[4], a[5], a[6], a[7] };
        *reinterpret_cast<f32x4*>(po)     = s0;
        *reinterpret_cast<f32x4*>(po + 4) = s1;
    }
}

extern "C" void kernel_launch(void* const* d_in, const int* in_sizes, int n_in,
                              void* d_out, int out_size, void* d_ws, size_t ws_size,
                              hipStream_t stream) {
    (void)in_sizes; (void)n_in; (void)out_size;
    const float* in   = (const float*)d_in[0];
    const float* kern = (const float*)d_in[1];
    float* out        = (float*)d_out;
    unsigned short* wt = (unsigned short*)d_ws;

    if (ws_size < (size_t)WELT_ * sizeof(unsigned short)) return; // need 663KB scratch

    conv4d_wprep<<<WELT_ / 256, 256, 0, stream>>>(kern, wt);
    conv4d_wmma<<<NBLK_, 512, 0, stream>>>(in, wt, out);
}
